// DiTBlock_6777458393115
// MI455X (gfx1250) — matrix operand order repaired and checked by reading, not compiled, not run on hardware
//
#include <hip/hip_runtime.h>
#include <cmath>

// ---------------------------------------------------------------------------
// DiT block for MI455X (gfx1250). All matmuls on v_wmma_f32_16x16x32_f16,
// f16 operands end-to-end on hot paths, fp32 accumulate, fp32 output.
// ---------------------------------------------------------------------------

typedef _Float16 f16;
typedef _Float16 v16h __attribute__((ext_vector_type(16)));
typedef _Float16 v8h  __attribute__((ext_vector_type(8)));
typedef _Float16 v4h  __attribute__((ext_vector_type(4)));
typedef float    v8f  __attribute__((ext_vector_type(8)));
typedef long long i64;

static constexpr int kB  = 4;
static constexpr int kS  = 1024;
static constexpr int kSC = 256;
static constexpr int kD  = 1024;
static constexpr int kH  = 16;
static constexpr int kHD = 64;
static constexpr int kFF = 4096;
static constexpr int k3D = 3 * kD;

#define LDPAD 40   // GEMM LDS tile row: 32 data + 8 pad halves (16B-aligned rows)
#define LDP2  72   // flash P tile row: 64 data + 8 pad halves (16B-aligned rows)

enum { EPI_F32 = 0, EPI_F16 = 1, EPI_RES = 2, EPI_GATED = 3, EPI_GELU_F16 = 4,
       EPI_QKV = 5, EPI_F16T = 6 };

__device__ __forceinline__ v16h combine8(v8h lo, v8h hi) {
  return __builtin_shufflevector(lo, hi, 0,1,2,3,4,5,6,7,8,9,10,11,12,13,14,15);
}

// A fragment 16x32 f16 from LDS tile (row-major, `stride` halves/row).
// Contiguous k order (matches B): lane hi=0 holds K=0..15 ; hi=1 holds K=16..31.
__device__ __forceinline__ v16h load_a_frag(const f16* As, int m0, int lane, int stride) {
  int l = lane & 15, hi = lane >> 4;
  const f16* p = As + (m0 + l) * stride + hi * 16;
  v8h lo = *(const v8h*)(p);
  v8h hh = *(const v8h*)(p + 8);
  return combine8(lo, hh);
}

// B fragment 32x16 f16 from LDS tile stored [n][k] (LDPAD halves/row).
// Contiguous k order (matches A): lanes 0-15 -> K=0..15, lanes 16-31 -> K=16..31, col = lane&15.
__device__ __forceinline__ v16h load_b_frag(const f16* Bs, int n0, int lane) {
  int l = lane & 15, hi = lane >> 4;
  const f16* p = Bs + (n0 + l) * LDPAD + hi * 16;
  v8h lo = *(const v8h*)(p);
  v8h hh = *(const v8h*)(p + 8);
  return combine8(lo, hh);
}

// ---------------------------------------------------------------------------
// fp32 -> f16 converter (weights, once per launch; n multiple of 1024)
// ---------------------------------------------------------------------------
__global__ __launch_bounds__(256)
void cvt_f16(const float* __restrict__ in, f16* __restrict__ out, int n)
{
  int i = (blockIdx.x * 256 + threadIdx.x) * 4;
  if (i < n) {
    float4 v = *(const float4*)(in + i);
    v4h o = { (f16)v.x, (f16)v.y, (f16)v.z, (f16)v.w };
    *(v4h*)(out + i) = o;
  }
}

// ---------------------------------------------------------------------------
// WMMA GEMM:  C[z][M,N] = A[z][M,K] * B[z][N,K]^T + bias
// A,B f16, row stride K. Block tile 128x128, 8 waves (4x2), wave tile 32x64
// = 2x4 WMMA 16x16x32 per 32-wide K step.
// EPI_QKV: writes Q cols scaled 1/8, K cols plain, V cols transposed per head
//          into Oaux as VT[b][h][hd][key] (so flash PV B-frags are contiguous).
// ---------------------------------------------------------------------------
template<int EPI>
__global__ __launch_bounds__(256)
void gemm_wmma(const f16* __restrict__ A, const f16* __restrict__ B,
               const float* __restrict__ bias, const float* __restrict__ res,
               const float* __restrict__ gate,
               float* __restrict__ Of32, f16* __restrict__ Of16,
               f16* __restrict__ Oaux,
               int M, int N, int K, i64 sAB, i64 sBB, i64 sCB)
{
  __shared__ f16 As[128 * LDPAD];
  __shared__ f16 Bs[128 * LDPAD];

  const int tid  = threadIdx.x;
  const int wave = tid >> 5;
  const int lane = tid & 31;
  const int l    = lane & 15, hic = lane >> 4;
  const int z     = blockIdx.z;
  const int mBase = blockIdx.y * 128;
  const int nBase = blockIdx.x * 128;
  const int wm = (wave & 3) * 32;
  const int wn = (wave >> 2) * 64;

  const f16* Ab = A + (i64)z * sAB;
  const f16* Bb = B + (i64)z * sBB;

  v8f zero = {};
  v8f acc[2][4];
  #pragma unroll
  for (int i = 0; i < 2; ++i)
    #pragma unroll
    for (int j = 0; j < 4; ++j) acc[i][j] = zero;

  for (int k0 = 0; k0 < K; k0 += 32) {
    // stage A tile 128x32 and B tile 128(n)x32(k), both f16 vector copies
    #pragma unroll
    for (int it = 0; it < 2; ++it) {
      int e = (it * 256 + tid) * 8;
      int r = e >> 5, c = e & 31;
      *(v8h*)(As + r * LDPAD + c) = *(const v8h*)(Ab + (i64)(mBase + r) * K + k0 + c);
      *(v8h*)(Bs + r * LDPAD + c) = *(const v8h*)(Bb + (i64)(nBase + r) * K + k0 + c);
    }
    if (k0 + 32 < K) {   // prefetch next K tile (global_prefetch_b8)
      int r = tid >> 1, sel = (tid & 1) * 16;
      __builtin_prefetch(Ab + (i64)(mBase + r) * K + k0 + 32 + sel, 0, 0);
      __builtin_prefetch(Bb + (i64)(nBase + r) * K + k0 + 32 + sel, 0, 0);
    }
    __syncthreads();

    v16h a0 = load_a_frag(As, wm,      lane, LDPAD);
    v16h a1 = load_a_frag(As, wm + 16, lane, LDPAD);
    #pragma unroll
    for (int j = 0; j < 4; ++j) {
      v16h b = load_b_frag(Bs, wn + j * 16, lane);
      acc[0][j] = __builtin_amdgcn_wmma_f32_16x16x32_f16(false, a0, false, b, (short)0, acc[0][j], false, false);
      acc[1][j] = __builtin_amdgcn_wmma_f32_16x16x32_f16(false, a1, false, b, (short)0, acc[1][j], false, false);
    }
    __syncthreads();
  }

  float gv = 0.0f;
  if (EPI == EPI_GATED) gv = tanhf(gate[0]);

  #pragma unroll
  for (int i = 0; i < 2; ++i) {
    #pragma unroll
    for (int j = 0; j < 4; ++j) {
      int col = nBase + wn + j * 16 + l;
      float bvv = bias ? bias[col] : 0.0f;
      #pragma unroll
      for (int r = 0; r < 8; ++r) {
        int row = mBase + wm + i * 16 + r + hic * 8;   // C layout: M = vgpr + 8*hi
        i64 idx = (i64)z * sCB + (i64)row * N + col;
        float v = acc[i][j][r] + bvv;
        if      (EPI == EPI_F32)   Of32[idx] = v;
        else if (EPI == EPI_F16)   Of16[idx] = (f16)v;
        else if (EPI == EPI_RES)   Of32[idx] = res[idx] + v;
        else if (EPI == EPI_GATED) Of32[idx] = res[idx] + gv * v;
        else if (EPI == EPI_GELU_F16) {
          float g = 0.5f * v * (1.0f + erff(v * 0.70710678118654752f)); // exact gelu
          Of16[idx] = (f16)g;
        }
        else if (EPI == EPI_QKV) {
          if (col < kD)            Of16[idx] = (f16)(v * 0.125f);   // Q * 1/sqrt(HD)
          else if (col < 2 * kD)   Of16[idx] = (f16)v;              // K
          else {                                                     // V transposed
            int bb = row >> 10, s = row & (kS - 1);                  // S = 1024
            int hh = (col - 2 * kD) >> 6, hd = (col - 2 * kD) & 63;  // HD = 64
            Oaux[((i64)(bb * kH + hh) * kHD + hd) * kS + s] = (f16)v;
          }
        }
        else if (EPI == EPI_F16T) {           // transposed store: [N, M] per batch
          Of16[(i64)z * sCB + (i64)col * M + row] = (f16)v;
        }
      }
    }
  }
}

// ---------------------------------------------------------------------------
// LayerNorm (D=1024) -> f16 output (A-operand for next GEMM)
// ---------------------------------------------------------------------------
__global__ __launch_bounds__(256)
void ln_f16(const float* __restrict__ x, const float* __restrict__ g,
            const float* __restrict__ be, f16* __restrict__ out)
{
  const int row = blockIdx.x;
  const int tid = threadIdx.x;
  const float* xp = x + (i64)row * kD;
  __shared__ float r0[8], r1[8];
  float s = 0.f, ss = 0.f;
  #pragma unroll
  for (int i = tid; i < kD; i += 256) { float v = xp[i]; s += v; ss += v * v; }
  #pragma unroll
  for (int off = 16; off >= 1; off >>= 1) { s += __shfl_xor(s, off, 32); ss += __shfl_xor(ss, off, 32); }
  if ((tid & 31) == 0) { r0[tid >> 5] = s; r1[tid >> 5] = ss; }
  __syncthreads();
  s = r0[tid & 7]; ss = r1[tid & 7];
  #pragma unroll
  for (int off = 4; off >= 1; off >>= 1) { s += __shfl_xor(s, off, 32); ss += __shfl_xor(ss, off, 32); }
  float mean = s * (1.0f / kD);
  float var  = ss * (1.0f / kD) - mean * mean;
  float inv  = rsqrtf(var + 1e-5f);
  #pragma unroll
  for (int i = tid; i < kD; i += 256) {
    float v = (xp[i] - mean) * inv * g[i] + be[i];
    out[(i64)row * kD + i] = (f16)v;
  }
}

// ---------------------------------------------------------------------------
// Flash self-attention, all-f16 operands: one wave per (b, h, 16-query tile).
// qkv f16 [B,S,3D] (Q pre-scaled by 1/8), VT f16 [B,H,HD,S]. out f16 [B,S,D].
// 64 keys per iteration: 16 WMMA + one row-max butterfly per 64 keys; softmax
// denominator reduction deferred to the end via corr-rescaled per-lane partials.
// ---------------------------------------------------------------------------
__global__ __launch_bounds__(32)
void flash_attn(const f16* __restrict__ qkv, const f16* __restrict__ VT,
                f16* __restrict__ out)
{
  const int lane = threadIdx.x;
  const int l = lane & 15, hic = lane >> 4;
  const int q0 = blockIdx.x * 16;
  const int h  = blockIdx.y;
  const int b  = blockIdx.z;
  __shared__ f16 Pl[16 * LDP2];

  // Q fragments (K-dim = HD = 64 -> two 16x32 A frags), pure vector loads
  const f16* qp = qkv + (i64)(b * kS + q0 + l) * k3D + h * kHD;
  v16h aq[2];
  #pragma unroll
  for (int ks = 0; ks < 2; ++ks)
    aq[ks] = combine8(*(const v8h*)(qp + ks * 32 + hic * 16),
                      *(const v8h*)(qp + ks * 32 + hic * 16 + 8));

  const f16* vtb = VT + (i64)(b * kH + h) * kHD * kS;

  float mrow[8];
  #pragma unroll
  for (int r = 0; r < 8; ++r) mrow[r] = -1e30f;
  v8f zero = {};
  v8f sacc = zero;          // per-lane partial softmax denominator (corr-rescaled)
  v8f o[4];
  #pragma unroll
  for (int j = 0; j < 4; ++j) o[j] = zero;

  for (int kt = 0; kt < kS; kt += 64) {
    // scores tile 16q x 64keys: B[k=hd][n=key], contiguous f16 from K rows
    v8f s[4];
    #pragma unroll
    for (int jt = 0; jt < 4; ++jt) s[jt] = zero;
    #pragma unroll
    for (int jt = 0; jt < 4; ++jt) {
      const f16* kp = qkv + (i64)(b * kS + kt + jt * 16 + l) * k3D + kD + h * kHD;
      #pragma unroll
      for (int ks = 0; ks < 2; ++ks) {
        v16h bk = combine8(*(const v8h*)(kp + ks * 32 + hic * 16),
                           *(const v8h*)(kp + ks * 32 + hic * 16 + 8));
        s[jt] = __builtin_amdgcn_wmma_f32_16x16x32_f16(false, aq[ks], false, bk, (short)0, s[jt], false, false);
      }
    }
    // online softmax: row = r + 8*hic lives on 16 lanes of one half-wave
    #pragma unroll
    for (int r = 0; r < 8; ++r) {
      float vmax = fmaxf(fmaxf(s[0][r], s[1][r]), fmaxf(s[2][r], s[3][r]));
      #pragma unroll
      for (int off = 8; off >= 1; off >>= 1) vmax = fmaxf(vmax, __shfl_xor(vmax, off, 32));
      float mnew = fmaxf(mrow[r], vmax);
      float corr = __expf(mrow[r] - mnew);
      float psum = 0.0f;
      #pragma unroll
      for (int jt = 0; jt < 4; ++jt) {
        float p = __expf(s[jt][r] - mnew);
        psum += p;
        Pl[(r + 8 * hic) * LDP2 + jt * 16 + l] = (f16)p;
      }
      sacc[r] = sacc[r] * corr + psum;        // deferred cross-lane reduction
      mrow[r] = mnew;
      #pragma unroll
      for (int j = 0; j < 4; ++j) o[j][r] *= corr;
    }
    __syncthreads();
    v16h ap0 = load_a_frag(Pl,      0, lane, LDP2);   // P keys kt+0..31
    v16h ap1 = load_a_frag(Pl + 32, 0, lane, LDP2);   // P keys kt+32..63
    #pragma unroll
    for (int j = 0; j < 4; ++j) {
      // PV B-frags: V^T[hd][key], key contiguous -> 16B vector loads
      const f16* p = vtb + (i64)(j * 16 + l) * kS + kt + hic * 16;
      v16h bv0 = combine8(*(const v8h*)(p),      *(const v8h*)(p + 8));
      v16h bv1 = combine8(*(const v8h*)(p + 32), *(const v8h*)(p + 40));
      o[j] = __builtin_amdgcn_wmma_f32_16x16x32_f16(false, ap0, false, bv0, (short)0, o[j], false, false);
      o[j] = __builtin_amdgcn_wmma_f32_16x16x32_f16(false, ap1, false, bv1, (short)0, o[j], false, false);
    }
    __syncthreads();
  }

  // final denominator: one 16-lane reduction per row
  float srow[8];
  #pragma unroll
  for (int r = 0; r < 8; ++r) {
    float s = sacc[r];
    #pragma unroll
    for (int off = 8; off >= 1; off >>= 1) s += __shfl_xor(s, off, 32);
    srow[r] = s;
  }

  #pragma unroll
  for (int j = 0; j < 4; ++j)
    #pragma unroll
    for (int r = 0; r < 8; ++r) {
      int row = r + 8 * hic;
      float v = o[j][r] / srow[r];
      out[(i64)(b * kS + q0 + row) * kD + h * kHD + j * 16 + l] = (f16)v;
    }
}

// ---------------------------------------------------------------------------
// Cross-attn masked softmax over SC=256 cols; scale 1/sqrt(D)=1/32. out f16.
// ---------------------------------------------------------------------------
__global__ __launch_bounds__(256)
void mask_softmax(const float* __restrict__ sc, const int* __restrict__ mask,
                  f16* __restrict__ out)
{
  const i64 rowi = blockIdx.x;          // b*S + q
  const i64 b = rowi / kS;
  const int tid = threadIdx.x;          // col
  float v = sc[rowi * kSC + tid] * 0.03125f;
  if (mask[b * kSC + tid] == 0) v = -10000.0f;
  __shared__ float red[8];
  float m = v;
  #pragma unroll
  for (int off = 16; off >= 1; off >>= 1) m = fmaxf(m, __shfl_xor(m, off, 32));
  if ((tid & 31) == 0) red[tid >> 5] = m;
  __syncthreads();
  m = red[tid & 7];
  #pragma unroll
  for (int off = 4; off >= 1; off >>= 1) m = fmaxf(m, __shfl_xor(m, off, 32));
  float e = __expf(v - m);
  float s = e;
  __syncthreads();
  #pragma unroll
  for (int off = 16; off >= 1; off >>= 1) s += __shfl_xor(s, off, 32);
  if ((tid & 31) == 0) red[tid >> 5] = s;
  __syncthreads();
  s = red[tid & 7];
  #pragma unroll
  for (int off = 4; off >= 1; off >>= 1) s += __shfl_xor(s, off, 32);
  out[rowi * kSC + tid] = (f16)(e / s);
}

// ---------------------------------------------------------------------------
extern "C" void kernel_launch(void* const* d_in, const int* in_sizes, int n_in,
                              void* d_out, int out_size, void* d_ws, size_t ws_size,
                              hipStream_t stream)
{
  (void)in_sizes; (void)n_in; (void)out_size; (void)ws_size;
  const float* x        = (const float*)d_in[0];
  const float* cond     = (const float*)d_in[1];
  const int*   cmask    = (const int*)  d_in[2];
  const float* ln1_g    = (const float*)d_in[3];
  const float* ln1_b    = (const float*)d_in[4];
  const float* sa_in_w  = (const float*)d_in[5];
  const float* sa_in_b  = (const float*)d_in[6];
  const float* sa_out_w = (const float*)d_in[7];
  const float* sa_out_b = (const float*)d_in[8];
  const float* ln2x_g   = (const float*)d_in[9];
  const float* ln2x_b   = (const float*)d_in[10];
  const float* ln2c_g   = (const float*)d_in[11];
  const float* ln2c_b   = (const float*)d_in[12];
  const float* wq  = (const float*)d_in[13];
  const float* bq  = (const float*)d_in[14];
  const float* wk  = (const float*)d_in[15];
  const float* bk  = (const float*)d_in[16];
  const float* wv  = (const float*)d_in[17];
  const float* bv  = (const float*)d_in[18];
  const float* wo  = (const float*)d_in[19];
  const float* bo  = (const float*)d_in[20];
  const float* gate= (const float*)d_in[21];
  const float* ln3_g = (const float*)d_in[22];
  const float* ln3_b = (const float*)d_in[23];
  const float* w1  = (const float*)d_in[24];
  const float* b1  = (const float*)d_in[25];
  const float* w2  = (const float*)d_in[26];
  const float* b2  = (const float*)d_in[27];

  char* ws = (char*)d_ws;
  const size_t MB = 1024 * 1024;
  f16*   qkv16   = (f16*)  (ws + 0);         // 24 MB [B,S,3D] f16 (Q pre-scaled)
  f16*   h16     = (f16*)  (ws + 0);         // reuse: 32 MB [B,S,FF] f16
  f16*   w_in16  = (f16*)  (ws + 32  * MB);  // 6 MB
  f16*   w_out16 = (f16*)  (ws + 38  * MB);  // 2 MB
  f16*   wq16    = (f16*)  (ws + 40  * MB);  // 2 MB
  f16*   wk16    = (f16*)  (ws + 42  * MB);  // 2 MB
  f16*   wv16    = (f16*)  (ws + 44  * MB);  // 2 MB
  f16*   wo16    = (f16*)  (ws + 46  * MB);  // 2 MB
  f16*   w1_16   = (f16*)  (ws + 48  * MB);  // 8 MB
  f16*   w2_16   = (f16*)  (ws + 56  * MB);  // 8 MB
  f16*   xs16    = (f16*)  (ws + 64  * MB);  // 8 MB LN outputs (reused 3x)
  f16*   sa16    = (f16*)  (ws + 72  * MB);  // 8 MB (later cq16 / cross16)
  f16*   cq16    = (f16*)  (ws + 72  * MB);
  f16*   cross16 = (f16*)  (ws + 72  * MB);
  float* x1      = (float*)(ws + 80  * MB);  // 16 MB
  float* x2      = (float*)(ws + 96  * MB);  // 16 MB
  f16*   cn16    = (f16*)  (ws + 112 * MB);  // 2 MB
  f16*   ck16    = (f16*)  (ws + 114 * MB);  // 2 MB
  f16*   cvT16   = (f16*)  (ws + 116 * MB);  // 2 MB [b][D][SC] (transposed)
  float* attnS   = (float*)(ws + 118 * MB);  // 4 MB [B,S,SC]
  f16*   attnP16 = (f16*)  (ws + 122 * MB);  // 2 MB
  f16*   vT16    = (f16*)  (ws + 124 * MB);  // 8 MB [B,H,HD,S]

  const i64 z0 = 0;
  // 0) weights -> f16 (read once; ~3us of HBM time at 23.3 TB/s)
  cvt_f16<<<(k3D*kD)/1024, 256, 0, stream>>>(sa_in_w,  w_in16,  k3D*kD);
  cvt_f16<<<(kD*kD)/1024,  256, 0, stream>>>(sa_out_w, w_out16, kD*kD);
  cvt_f16<<<(kD*kD)/1024,  256, 0, stream>>>(wq, wq16, kD*kD);
  cvt_f16<<<(kD*kD)/1024,  256, 0, stream>>>(wk, wk16, kD*kD);
  cvt_f16<<<(kD*kD)/1024,  256, 0, stream>>>(wv, wv16, kD*kD);
  cvt_f16<<<(kD*kD)/1024,  256, 0, stream>>>(wo, wo16, kD*kD);
  cvt_f16<<<(kFF*kD)/1024, 256, 0, stream>>>(w1, w1_16, kFF*kD);
  cvt_f16<<<(kD*kFF)/1024, 256, 0, stream>>>(w2, w2_16, kD*kFF);

  // 1) xs = LN1(x)
  ln_f16<<<kB * kS, 256, 0, stream>>>(x, ln1_g, ln1_b, xs16);
  // 2) qkv = xs @ sa_in_w^T + b (f16; Q scaled 1/8, V written transposed to vT16)
  gemm_wmma<EPI_QKV><<<dim3(k3D/128, (kB*kS)/128, 1), 256, 0, stream>>>(
      xs16, w_in16, sa_in_b, nullptr, nullptr,
      nullptr, qkv16, vT16, kB*kS, k3D, kD, z0, z0, z0);
  // 3) self-attention (flash, WMMA, all f16 operands)
  flash_attn<<<dim3(kS/16, kH, kB), 32, 0, stream>>>(qkv16, vT16, sa16);
  // 4) x1 = x + sa @ sa_out_w^T + b
  gemm_wmma<EPI_RES><<<dim3(kD/128, (kB*kS)/128, 1), 256, 0, stream>>>(
      sa16, w_out16, sa_out_b, x, nullptr,
      x1, nullptr, nullptr, kB*kS, kD, kD, z0, z0, z0);
  // 5) LN2x(x1), LN2c(cond)
  ln_f16<<<kB * kS, 256, 0, stream>>>(x1, ln2x_g, ln2x_b, xs16);
  ln_f16<<<kB * kSC, 256, 0, stream>>>(cond, ln2c_g, ln2c_b, cn16);
  // 6) cq / ck projections (f16 out); cv projected with transposed store
  gemm_wmma<EPI_F16><<<dim3(kD/128, (kB*kS)/128, 1), 256, 0, stream>>>(
      xs16, wq16, bq, nullptr, nullptr,
      nullptr, cq16, nullptr, kB*kS, kD, kD, z0, z0, z0);
  gemm_wmma<EPI_F16><<<dim3(kD/128, (kB*kSC)/128, 1), 256, 0, stream>>>(
      cn16, wk16, bk, nullptr, nullptr,
      nullptr, ck16, nullptr, kB*kSC, kD, kD, z0, z0, z0);
  gemm_wmma<EPI_F16T><<<dim3(kD/128, kSC/128, kB), 256, 0, stream>>>(
      cn16, wv16, bv, nullptr, nullptr,
      nullptr, cvT16, nullptr, kSC, kD, kD, (i64)kSC*kD, z0, (i64)kSC*kD);
  // 7) scores[b] = cq[b] @ ck[b]^T  (batched; B = ck f16 [N=SC, K=D])
  gemm_wmma<EPI_F32><<<dim3(kSC/128, kS/128, kB), 256, 0, stream>>>(
      cq16, ck16, nullptr, nullptr, nullptr,
      attnS, nullptr, nullptr, kS, kSC, kD, (i64)kS*kD, (i64)kSC*kD, (i64)kS*kSC);
  // 8) masked softmax over SC
  mask_softmax<<<kB * kS, 256, 0, stream>>>(attnS, cmask, attnP16);
  // 9) cross[b] = attnP[b] @ cvT[b]^T  (cvT is [N=D, K=SC], contiguous)
  gemm_wmma<EPI_F16><<<dim3(kD/128, kS/128, kB), 256, 0, stream>>>(
      attnP16, cvT16, nullptr, nullptr, nullptr,
      nullptr, cross16, nullptr, kS, kD, kSC, (i64)kS*kSC, (i64)kSC*kD, (i64)kS*kD);
  // 10) x2 = x1 + tanh(gate)*(cross @ wo^T + bo)
  gemm_wmma<EPI_GATED><<<dim3(kD/128, (kB*kS)/128, 1), 256, 0, stream>>>(
      cross16, wo16, bo, x1, gate,
      x2, nullptr, nullptr, kB*kS, kD, kD, z0, z0, z0);
  // 11) LN3 + FFN
  ln_f16<<<kB * kS, 256, 0, stream>>>(x2, ln3_g, ln3_b, xs16);
  gemm_wmma<EPI_GELU_F16><<<dim3(kFF/128, (kB*kS)/128, 1), 256, 0, stream>>>(
      xs16, w1_16, b1, nullptr, nullptr,
      nullptr, h16, nullptr, kB*kS, kFF, kD, z0, z0, z0);
  gemm_wmma<EPI_RES><<<dim3(kD/128, (kB*kS)/128, 1), 256, 0, stream>>>(
      h16, w2_16, b2, x2, nullptr,
      (float*)d_out, nullptr, nullptr, kB*kS, kD, kFF, z0, z0, z0);
}
